// GridSelfAttention_33913061769434
// MI455X (gfx1250) — matrix operand order repaired and checked by reading, not compiled, not run on hardware
//
#include <hip/hip_runtime.h>
#include <hip/hip_bf16.h>

typedef __bf16 bf16;
typedef __attribute__((ext_vector_type(16))) __bf16 v16bf;
typedef __attribute__((ext_vector_type(8)))  float  v8f;

union bfrag { v16bf v; uint4 q[2]; };

#define NSEQ 320
#define CCH  128
#define NH   4
#define DH   32
#define MTOT (NSEQ*NSEQ)   // 102400

static __device__ __forceinline__ v8f wmma_bf16(const bfrag a, const bfrag b, v8f c) {
  // D = A(16x32 bf16) * B(32x16 bf16) + C(16x16 f32)
  return __builtin_amdgcn_wmma_f32_16x16x32_bf16(false, a.v, false, b.v, (short)0, c, false, false);
}

// ---------------------------------------------------------------------------
// Kernel 0: convert weights to bf16 in WMMA-B fragment-ready swizzled layout.
// B fragment layout (32x16, 8 VGPRs, wave32): lane l element i corresponds to
//   row K = (i&7) + 16*(i>=8) + 8*(l>=16), col N = l%16  -> store 16 contiguous bf16 per lane.
// wcat columns: [0,128)=wq*scale  [128,256)=wk  [256,384)=wv
//               [384,512)=w_gate  [512,528)=w_pair_bias (padded to 16)
// ---------------------------------------------------------------------------
__global__ void prep_kernel(const float* __restrict__ wq, const float* __restrict__ wk,
                            const float* __restrict__ wv, const float* __restrict__ wg,
                            const float* __restrict__ wpb, const float* __restrict__ wo,
                            bf16* __restrict__ wcat, bf16* __restrict__ woT) {
  const float qscale = 0.17677669529663687f; // 32^-0.5
  const int tid = blockIdx.x * blockDim.x + threadIdx.x;
  const int stride = gridDim.x * blockDim.x;
  for (int e = tid; e < 4*33*512; e += stride) {
    const int i  = e & 15;
    const int l  = (e >> 4) & 31;
    const int nt = (e >> 9) % 33;
    const int kt = e / (33*512);
    const int col = nt*16 + (l & 15);
    const int k   = kt*32 + (i & 7) + ((i >= 8) ? 16 : 0) + ((l >= 16) ? 8 : 0);
    float val;
    if (col < 128)      val = wq[k*128 + col] * qscale;
    else if (col < 256) val = wk[k*128 + (col-128)];
    else if (col < 384) val = wv[k*128 + (col-256)];
    else if (col < 512) val = wg[k*128 + (col-384)];
    else { const int h = col - 512; val = (h < 4) ? wpb[k*4 + h] : 0.0f; }
    wcat[e] = (bf16)val;
  }
  // w_out applied as y @ w_out.T  ->  B[k][o] = w_out[o][k]
  for (int e = tid; e < 4*8*512; e += stride) {
    const int i  = e & 15;
    const int l  = (e >> 4) & 31;
    const int nt = (e >> 9) & 7;
    const int kt = e >> 12;
    const int o  = nt*16 + (l & 15);
    const int k  = kt*32 + (i & 7) + ((i >= 8) ? 16 : 0) + ((l >= 16) ? 8 : 0);
    woT[e] = (bf16)wo[o*128 + k];
  }
}

// ---------------------------------------------------------------------------
// Kernel 1: LayerNorm over C=128, one wave per row, output bf16.
// ---------------------------------------------------------------------------
__global__ void ln_kernel(const float* __restrict__ act, const float* __restrict__ gamma,
                          const float* __restrict__ beta, bf16* __restrict__ x) {
  const int row  = blockIdx.x * 8 + (threadIdx.x >> 5);
  const int lane = threadIdx.x & 31;
  const float4 p = ((const float4*)(act + (size_t)row * 128))[lane];
  float s  = p.x + p.y + p.z + p.w;
  float s2 = p.x*p.x + p.y*p.y + p.z*p.z + p.w*p.w;
  #pragma unroll
  for (int off = 16; off > 0; off >>= 1) {
    s  += __shfl_xor(s,  off, 32);
    s2 += __shfl_xor(s2, off, 32);
  }
  const float mean = s * (1.0f/128.0f);
  const float var  = s2 * (1.0f/128.0f) - mean*mean;
  const float inv  = rsqrtf(var + 1e-5f);
  const float4 g  = ((const float4*)gamma)[lane];
  const float4 bt = ((const float4*)beta)[lane];
  union { bf16 h[4]; ushort4 u; } o;
  o.h[0] = (bf16)((p.x - mean)*inv*g.x + bt.x);
  o.h[1] = (bf16)((p.y - mean)*inv*g.y + bt.y);
  o.h[2] = (bf16)((p.z - mean)*inv*g.z + bt.z);
  o.h[3] = (bf16)((p.w - mean)*inv*g.w + bt.w);
  ((ushort4*)(x + (size_t)row * 128))[lane] = o.u;
}

// ---------------------------------------------------------------------------
// Kernel 2: fused projection GEMM  x[102400,128] @ Wcat[128,528]
// A fragment (16x32 bf16): lane<16 -> row=lane, K {0..7,16..23};
//                          lane>=16 -> row=lane-16, K {8..15,24..31}.
// C/D fragment (16x16 f32): vgpr r -> row r + 8*(lane>=16), col lane%16.
// ---------------------------------------------------------------------------
__global__ void proj_kernel(const bf16* __restrict__ x, const bf16* __restrict__ wcat,
                            bf16* __restrict__ q, bf16* __restrict__ k, bf16* __restrict__ vT,
                            bf16* __restrict__ gate, float* __restrict__ bias) {
  const int lane = threadIdx.x & 31;
  const int w    = threadIdx.x >> 5;
  const int m0   = (blockIdx.x * 8 + w) * 16;
  const int nt   = blockIdx.y;
  const int lr   = lane & 15;
  const bool hi  = lane >= 16;

  bfrag a[4];
  const bf16* arow = x + (size_t)(m0 + lr) * 128;
  #pragma unroll
  for (int kt = 0; kt < 4; kt++) {
    const uint4* p = (const uint4*)(arow + kt*32 + (hi ? 8 : 0));
    a[kt].q[0] = p[0];           // K base .. +7
    a[kt].q[1] = p[2];           // K base+16 .. +23
  }
  v8f acc = {};
  #pragma unroll
  for (int kt = 0; kt < 4; kt++) {
    bfrag bw;
    const uint4* p = (const uint4*)(wcat + ((kt*33 + nt) << 9) + (lane << 4));
    bw.q[0] = p[0];
    bw.q[1] = p[1];
    acc = wmma_bf16(a[kt], bw, acc);
  }
  const int col = nt*16 + lr;
  #pragma unroll
  for (int r = 0; r < 8; r++) {
    const int m = m0 + r + (hi ? 8 : 0);
    const float vvv = acc[r];
    if (col < 128) {
      q[(size_t)m*128 + col] = (bf16)vvv;
    } else if (col < 256) {
      k[(size_t)m*128 + (col - 128)] = (bf16)vvv;
    } else if (col < 384) {
      const int c2 = col - 256;
      const int hh = c2 >> 5, dd = c2 & 31;
      const int bb = m / NSEQ, kk = m % NSEQ;   // m = b*320 + k_
      vT[((size_t)(bb*NH + hh)*32 + dd)*NSEQ + kk] = (bf16)vvv;
    } else if (col < 512) {
      gate[(size_t)m*128 + (col - 384)] = (bf16)(1.0f / (1.0f + __expf(-vvv)));
    } else {
      const int hh = col - 512;
      if (hh < 4) bias[(size_t)hh*MTOT + m] = vvv;   // bias[h, q, k] with m=q*320+k
    }
  }
}

// ---------------------------------------------------------------------------
// Kernel 3: attention per (b, h, 16-query tile).
//   phase1: logits strip [16 x 320] via 20 WMMAs (8 waves), +bias, mask
//   phase2: fp32 softmax in LDS, probs -> bf16 LDS
//   phase3: wa = P @ V (2 waves x 10 WMMAs), * gate, -> wag bf16
// ---------------------------------------------------------------------------
__global__ void attn_kernel(const bf16* __restrict__ q, const bf16* __restrict__ k,
                            const bf16* __restrict__ vT, const bf16* __restrict__ gate,
                            const float* __restrict__ bias, const int* __restrict__ pmask,
                            bf16* __restrict__ wag) {
  __shared__ float sL[16][328];
  __shared__ __align__(16) bf16 sP[16][336];
  __shared__ float sRed[16][16];

  const int qt = blockIdx.x;
  const int h  = blockIdx.y;
  const int b  = blockIdx.z;
  const int lane = threadIdx.x & 31;
  const int w    = threadIdx.x >> 5;
  const int lr   = lane & 15;
  const bool hi  = lane >= 16;

  // phase 1: QK^T tiles (q already carries 1/sqrt(D) scale)
  bfrag aq;
  {
    const bf16* base = q + (size_t)(b*NSEQ + qt*16 + lr) * 128 + h*32 + (hi ? 16 : 0);
    const uint4* p = (const uint4*)base;
    aq.q[0] = p[0];
    aq.q[1] = p[1];
  }
  for (int kt = w; kt < 20; kt += 8) {
    const int kc = kt*16 + lr;
    bfrag bk;   // B[d][kc] = k[b*320+kc][h*32+d], contiguous in d
    {
      const bf16* base = k + (size_t)(b*NSEQ + kc) * 128 + h*32 + (hi ? 16 : 0);
      const uint4* p = (const uint4*)base;
      bk.q[0] = p[0];
      bk.q[1] = p[1];
    }
    v8f acc = {};
    acc = wmma_bf16(aq, bk, acc);
    const int mok = (pmask[kc*NSEQ + b] > 0);   // pair_mask.T[b, kc]
    #pragma unroll
    for (int r = 0; r < 8; r++) {
      const int rr = r + (hi ? 8 : 0);
      const int qg = qt*16 + rr;
      const float val = acc[r] + bias[(size_t)h*MTOT + qg*NSEQ + kc];
      sL[rr][kc] = mok ? val : -1e9f;
    }
  }
  __syncthreads();

  // phase 2: rowwise softmax; 16 threads per row, 20 columns each
  {
    const int r  = threadIdx.x >> 4;
    const int g  = threadIdx.x & 15;
    const int c0 = g * 20;
    float mx = -3.0e38f;
    for (int j = 0; j < 20; j++) mx = fmaxf(mx, sL[r][c0+j]);
    sRed[r][g] = mx;
    __syncthreads();
    float rm = -3.0e38f;
    for (int j = 0; j < 16; j++) rm = fmaxf(rm, sRed[r][j]);
    __syncthreads();
    float ls = 0.0f;
    for (int j = 0; j < 20; j++) {
      const float p = __expf(sL[r][c0+j] - rm);
      sL[r][c0+j] = p;
      ls += p;
    }
    sRed[r][g] = ls;
    __syncthreads();
    float tot = 0.0f;
    for (int j = 0; j < 16; j++) tot += sRed[r][j];
    const float inv = 1.0f / tot;
    for (int j = 0; j < 20; j++) sP[r][c0+j] = (bf16)(sL[r][c0+j] * inv);
  }
  __syncthreads();

  // phase 3: wa = P @ V   (waves 0,1; each a 16x16 tile of [16 x 32])
  if (w < 2) {
    const int n0 = w * 16;
    const int d  = n0 + lr;
    v8f acc = {};
    for (int kt = 0; kt < 10; kt++) {
      bfrag ap, bv;
      {
        const uint4* p = (const uint4*)(&sP[lr][kt*32 + (hi ? 16 : 0)]);
        ap.q[0] = p[0];
        ap.q[1] = p[1];
      }
      {
        const bf16* base = vT + ((size_t)((b*NH + h)*32 + d)) * NSEQ + kt*32 + (hi ? 16 : 0);
        const uint4* p = (const uint4*)base;
        bv.q[0] = p[0];
        bv.q[1] = p[1];
      }
      acc = wmma_bf16(ap, bv, acc);
    }
    const int c = h*32 + d;
    #pragma unroll
    for (int r = 0; r < 8; r++) {
      const int m = b*NSEQ + qt*16 + r + (hi ? 8 : 0);
      const float g = (float)gate[(size_t)m*128 + c];
      wag[(size_t)m*128 + c] = (bf16)(acc[r] * g);
    }
  }
}

// ---------------------------------------------------------------------------
// Kernel 4: out = (wa*gate) @ w_out.T  -> fp32
// ---------------------------------------------------------------------------
__global__ void out_kernel(const bf16* __restrict__ wag, const bf16* __restrict__ woT,
                           float* __restrict__ out) {
  const int lane = threadIdx.x & 31;
  const int w    = threadIdx.x >> 5;
  const int m0   = (blockIdx.x * 8 + w) * 16;
  const int nt   = blockIdx.y;
  const int lr   = lane & 15;
  const bool hi  = lane >= 16;

  bfrag a[4];
  const bf16* arow = wag + (size_t)(m0 + lr) * 128;
  #pragma unroll
  for (int kt = 0; kt < 4; kt++) {
    const uint4* p = (const uint4*)(arow + kt*32 + (hi ? 8 : 0));
    a[kt].q[0] = p[0];
    a[kt].q[1] = p[2];
  }
  v8f acc = {};
  #pragma unroll
  for (int kt = 0; kt < 4; kt++) {
    bfrag bw;
    const uint4* p = (const uint4*)(woT + ((kt*8 + nt) << 9) + (lane << 4));
    bw.q[0] = p[0];
    bw.q[1] = p[1];
    acc = wmma_bf16(a[kt], bw, acc);
  }
  const int col = nt*16 + lr;
  #pragma unroll
  for (int r = 0; r < 8; r++) {
    out[(size_t)(m0 + r + (hi ? 8 : 0))*128 + col] = acc[r];
  }
}

// ---------------------------------------------------------------------------
extern "C" void kernel_launch(void* const* d_in, const int* in_sizes, int n_in,
                              void* d_out, int out_size, void* d_ws, size_t ws_size,
                              hipStream_t stream) {
  (void)in_sizes; (void)n_in; (void)out_size; (void)ws_size;
  const float* act   = (const float*)d_in[0];
  const int*   pmask = (const int*)  d_in[1];
  const float* lng   = (const float*)d_in[2];
  const float* lnb   = (const float*)d_in[3];
  const float* wpb   = (const float*)d_in[4];
  const float* wq    = (const float*)d_in[5];
  const float* wk    = (const float*)d_in[6];
  const float* wv    = (const float*)d_in[7];
  const float* wg    = (const float*)d_in[8];
  const float* wo    = (const float*)d_in[9];
  float* out = (float*)d_out;

  char* ws = (char*)d_ws;
  size_t off = 0;
  auto alloc = [&](size_t bytes) -> char* {
    char* p = ws + off;
    off += (bytes + 255) & ~(size_t)255;
    return p;
  };
  bf16*  x    = (bf16*) alloc((size_t)MTOT * 128 * sizeof(bf16));
  bf16*  qb   = (bf16*) alloc((size_t)MTOT * 128 * sizeof(bf16));
  bf16*  kb   = (bf16*) alloc((size_t)MTOT * 128 * sizeof(bf16));
  bf16*  vT   = (bf16*) alloc((size_t)MTOT * 128 * sizeof(bf16));
  bf16*  gate = (bf16*) alloc((size_t)MTOT * 128 * sizeof(bf16));
  bf16*  wag  = (bf16*) alloc((size_t)MTOT * 128 * sizeof(bf16));
  float* bias = (float*)alloc((size_t)NH * MTOT * sizeof(float));
  bf16*  wcat = (bf16*) alloc((size_t)4 * 33 * 512 * sizeof(bf16));
  bf16*  woT  = (bf16*) alloc((size_t)4 * 8 * 512 * sizeof(bf16));

  prep_kernel<<<64, 256, 0, stream>>>(wq, wk, wv, wg, wpb, wo, wcat, woT);
  ln_kernel<<<MTOT/8, 256, 0, stream>>>(act, lng, lnb, x);
  proj_kernel<<<dim3(MTOT/128, 33), 256, 0, stream>>>(x, wcat, qb, kb, vT, gate, bias);
  attn_kernel<<<dim3(NSEQ/16, NH, NSEQ), 256, 0, stream>>>(qb, kb, vT, gate, bias, pmask, wag);
  out_kernel<<<dim3(MTOT/128, 8), 256, 0, stream>>>(wag, woT, out);
}
